// TGNRecommender_23502061044522
// MI455X (gfx1250) — matrix operand order repaired and checked by reading, not compiled, not run on hardware
//
#include <hip/hip_runtime.h>
#include <math.h>

// ---------------------------------------------------------------------------
// Types for WMMA fragments (trivially-copyable ext-vectors so unions are legal)
// ---------------------------------------------------------------------------
typedef __bf16 v16bf __attribute__((ext_vector_type(16)));
typedef float  v8f   __attribute__((ext_vector_type(8)));
typedef int    i32x4 __attribute__((ext_vector_type(4)));
typedef int    i32x2 __attribute__((ext_vector_type(2)));

#define HID 128
#define HD  256   // HEADS*OUT
#define OUTC 64
#define HEADS 4

// A fragment (16x32 bf16, per-lane): two 8-elem contiguous runs, second at +16
__device__ inline v16bf load_fragA(const __bf16* base) {
    union { v16bf v; i32x4 q[2]; } u;
    u.q[0] = *(const i32x4*)(base);
    u.q[1] = *(const i32x4*)(base + 16);
    return u.v;
}
// B fragment (32x16 bf16, per-lane): two 8-elem contiguous runs, second at +16
__device__ inline v16bf load_fragB(const __bf16* base) {
    union { v16bf v; i32x4 q[2]; } u;
    u.q[0] = *(const i32x4*)(base);
    u.q[1] = *(const i32x4*)(base + 16);
    return u.v;
}

// CDNA5 async copy: global -> LDS, 16 bytes, tracked by ASYNCcnt.
__device__ inline void async_ld_b128(void* lds_dst, const void* gsrc) {
    unsigned lds_off = (unsigned)(size_t)lds_dst;
    unsigned long long ga = (unsigned long long)(size_t)gsrc;
    asm volatile("global_load_async_to_lds_b128 %0, %1, off"
                 :: "v"(lds_off), "v"(ga)
                 : "memory");
}
__device__ inline void wait_async0() {
    asm volatile("s_wait_asynccnt 0x0" ::: "memory");
}

__device__ inline void atomicMaxFloat(float* addr, float val) {
    unsigned int* ua = (unsigned int*)addr;
    unsigned int old = *ua;
    while (val > __uint_as_float(old)) {
        unsigned int assumed = old;
        old = atomicCAS(ua, assumed, __float_as_uint(val));
        if (old == assumed) break;
    }
}

// ---------------------------------------------------------------------------
// Prep: transpose fp32 [K,Nc] -> bf16 [Nc,K] so B fragments are contiguous
// ---------------------------------------------------------------------------
__global__ void k_transpose_cvt(const float* __restrict__ src,
                                __bf16* __restrict__ dst, int K, int Nc) {
    int i = blockIdx.x * blockDim.x + threadIdx.x;
    if (i < K * Nc) {
        int k = i / Nc, n = i % Nc;
        dst[(size_t)n * K + k] = (__bf16)src[i];
    }
}

__global__ void k_init(float* __restrict__ smax, float* __restrict__ denom, int n4) {
    int i = blockIdx.x * blockDim.x + threadIdx.x;
    if (i < n4) { smax[i] = -3.402823466e38f; denom[i] = 0.0f; }
}

// h = memory + x, converted to bf16
__global__ void k_addcvt(const float* __restrict__ mem, const float* __restrict__ x,
                         __bf16* __restrict__ hb, int n) {
    int i = blockIdx.x * blockDim.x + threadIdx.x;
    if (i < n) hb[i] = (__bf16)(mem[i] + x[i]);
}

// ---------------------------------------------------------------------------
// Node projections: one wave per (16-row node tile, matrix). The wave holds all
// four A K-step fragments in registers and sweeps the 16 column tiles, so the
// A tile is loaded once instead of 16x.  Wt layout: [4][256][128] bf16.
// ---------------------------------------------------------------------------
__global__ __launch_bounds__(32)
void k_node_gemm(const __bf16* __restrict__ hb, const __bf16* __restrict__ Wt,
                 const float* __restrict__ bq, const float* __restrict__ bk,
                 const float* __restrict__ bv, const float* __restrict__ bs,
                 __bf16* __restrict__ qb, __bf16* __restrict__ kb,
                 __bf16* __restrict__ vb, float* __restrict__ acc, int N) {
    const int lane = threadIdx.x;
    const int tm   = blockIdx.x;
    const int mat  = blockIdx.y;         // 0=q 1=k 2=v 3=skip
    const int m    = lane & 15;          // A row / D col within tile
    const int kg   = lane >> 4;          // lane-half selector
    const int arow = min(tm * 16 + m, N - 1);

    const __bf16* Arow = hb + (size_t)arow * HID + kg * 8;
    v16bf a[4];
    #pragma unroll
    for (int k = 0; k < 4; k++) a[k] = load_fragA(Arow + k * 32);

    const float* bias = (mat == 0) ? bq : (mat == 1) ? bk : (mat == 2) ? bv : bs;
    __bf16* tgt = (mat == 0) ? qb : (mat == 1) ? kb : vb;
    const __bf16* Wm = Wt + (size_t)mat * (HD * HID);
    const int rowbase = tm * 16 + kg * 8;

    for (int nt = 0; nt < 16; nt++) {
        const int ncol = nt * 16 + m;
        const __bf16* Brow = Wm + (size_t)ncol * HID + kg * 8;
        v8f c = {};
        #pragma unroll
        for (int k = 0; k < 4; k++) {
            v16bf b = load_fragB(Brow + k * 32);
            c = __builtin_amdgcn_wmma_f32_16x16x32_bf16(false, a[k], false, b,
                                                        (short)0, c, false, false);
        }
        const float bcol = bias[ncol];
        if (mat == 3) {
            #pragma unroll
            for (int r = 0; r < 8; r++) {
                int row = min(rowbase + r, N - 1);
                acc[(size_t)row * HD + ncol] = c[r] + bcol;
            }
        } else {
            #pragma unroll
            for (int r = 0; r < 8; r++) {
                int row = min(rowbase + r, N - 1);
                tgt[(size_t)row * HD + ncol] = (__bf16)(c[r] + bcol);
            }
        }
    }
}

// ---------------------------------------------------------------------------
// Per-edge attention scores + running segment-max (one thread per edge*head)
// ---------------------------------------------------------------------------
__global__ void k_scores(const long long* __restrict__ ei,
                         const __bf16* __restrict__ qb, const __bf16* __restrict__ kb,
                         float* __restrict__ scores, float* __restrict__ smax, int E) {
    int t = blockIdx.x * blockDim.x + threadIdx.x;
    if (t >= E * HEADS) return;
    int e = t >> 2, hd = t & 3;
    long long s = ei[e], d = ei[E + e];
    const __bf16* qp = qb + (size_t)d * HD + hd * OUTC;
    const __bf16* kp = kb + (size_t)s * HD + hd * OUTC;
    float acc = 0.0f;
    #pragma unroll
    for (int i = 0; i < OUTC; i += 8) {
        union { i32x4 q; __bf16 h[8]; } uq, uk;
        uq.q = *(const i32x4*)(qp + i);
        uk.q = *(const i32x4*)(kp + i);
        #pragma unroll
        for (int j = 0; j < 8; j++) acc += (float)uq.h[j] * (float)uk.h[j];
    }
    acc *= 0.125f;  // 1/sqrt(64)
    scores[t] = acc;
    atomicMaxFloat(&smax[(size_t)d * HEADS + hd], acc);
}

__global__ void k_expnorm(const long long* __restrict__ ei,
                          float* __restrict__ scores, const float* __restrict__ smax,
                          float* __restrict__ denom, int E) {
    int t = blockIdx.x * blockDim.x + threadIdx.x;
    if (t >= E * HEADS) return;
    int e = t >> 2, hd = t & 3;
    long long d = ei[E + e];
    float ex = __expf(scores[t] - smax[(size_t)d * HEADS + hd]);
    scores[t] = ex;
    atomicAdd(&denom[(size_t)d * HEADS + hd], ex);
}

// scatter agg: acc[dst] += alpha * v[src]; one thread handles 4 dims
__global__ void k_scatter_v(const long long* __restrict__ ei,
                            const __bf16* __restrict__ vb,
                            const float* __restrict__ scores,
                            const float* __restrict__ denom,
                            float* __restrict__ acc, int E) {
    int t = blockIdx.x * blockDim.x + threadIdx.x;
    if (t >= E * (HD / 4)) return;
    int e    = t >> 6;
    int dim4 = (t & 63) * 4;
    int hd   = dim4 >> 6;
    long long s = ei[e], d = ei[E + e];
    float alpha = scores[(size_t)e * HEADS + hd] /
                  (denom[(size_t)d * HEADS + hd] + 1e-16f);
    union { i32x2 q; __bf16 h[4]; } u;
    u.q = *(const i32x2*)(vb + (size_t)s * HD + dim4);
    float* ap = acc + (size_t)d * HD + dim4;
    #pragma unroll
    for (int j = 0; j < 4; j++) atomicAdd(ap + j, alpha * (float)u.h[j]);
}

__global__ void k_cvt_out(const float* __restrict__ acc, __bf16* __restrict__ outb, int n) {
    int i = blockIdx.x * blockDim.x + threadIdx.x;
    if (i < n) outb[i] = (__bf16)acc[i];
}

// ---------------------------------------------------------------------------
// Fused per-edge rating MLP.
//   Block = 4 waves = 64 edges (16 per wave), grid = E/64.
//   W2t (16KB) staged into LDS once; W1t consumed as double-buffered 8KB
//   K-slices staged with global_load_async_to_lds_b128 (ASYNCcnt) so the
//   copy of slice k+1 overlaps the 32 WMMAs on slice k.
//   Layer1 (512->128): 16 K-steps x 8 N-tiles; ReLU -> LDS relay ->
//   Layer2 (128->64): 16 WMMA; ReLU -> LDS -> Layer3 (64->1) dot + sigmoid.
// ---------------------------------------------------------------------------
__global__ __launch_bounds__(128)
void k_edge_mlp(const long long* __restrict__ ei, const __bf16* __restrict__ outb,
                const __bf16* __restrict__ W1t, const float* __restrict__ b1,
                const __bf16* __restrict__ W2t, const float* __restrict__ b2,
                const float* __restrict__ W3, const float* __restrict__ b3,
                float* __restrict__ out, int E) {
    __shared__ __bf16 w1s[2][128][32];   // double-buffered W1 K-slice [n][k]
    __shared__ __bf16 w2s[64][128];      // whole W2t                 [n][k]
    __shared__ __bf16 relay1[4][16][128];
    __shared__ __bf16 relay2[4][16][64];

    const int tid   = threadIdx.x;
    const int lane  = tid & 31;
    const int wv    = tid >> 5;           // wave id 0..3
    const int m     = lane & 15;
    const int kg    = lane >> 4;
    const int ebase = blockIdx.x * 64 + wv * 16;
    const int er    = min(ebase + m, E - 1);
    const long long s = ei[er];
    const long long d = ei[E + er];
    const __bf16* srcRow = outb + (size_t)s * HD;
    const __bf16* dstRow = outb + (size_t)d * HD;

    // ---- prologue: stage W2t (each thread: 128B) and W1 slice 0 ----
    {
        const int n = tid >> 1, half = tid & 1;
        const __bf16* g = W2t + (size_t)n * 128 + half * 64;
        __bf16* l = &w2s[n][half * 64];
        #pragma unroll
        for (int j = 0; j < 8; j++) async_ld_b128(l + j * 8, g + j * 8);
    }
    {
        const __bf16* g = W1t + (size_t)tid * 512;   // slice 0: k in [0,32)
        __bf16* l = &w1s[0][tid][0];
        #pragma unroll
        for (int j = 0; j < 4; j++) async_ld_b128(l + j * 8, g + j * 8);
    }
    wait_async0();
    __syncthreads();

    // ---- layer 1: [16,512] x [512,128], W1 slices double-buffered ----
    v8f acc1[8];
    #pragma unroll
    for (int i = 0; i < 8; i++) acc1[i] = (v8f){};

    for (int idx = 0; idx < 16; idx++) {
        const int k0 = idx * 32;
        // kick off async copy of the next slice into the other buffer
        if (idx < 15) {
            const __bf16* g = W1t + (size_t)tid * 512 + (k0 + 32);
            __bf16* l = &w1s[(idx + 1) & 1][tid][0];
            #pragma unroll
            for (int j = 0; j < 4; j++) async_ld_b128(l + j * 8, g + j * 8);
        }
        // A fragment gathered from outb[src]|outb[dst]
        const int f0 = k0 + kg * 8;      // 32-chunk never crosses src/dst split
        const __bf16* ap = (f0 < HD) ? (srcRow + f0) : (dstRow + (f0 - HD));
        v16bf a = load_fragA(ap);
        const __bf16* wbuf = &w1s[idx & 1][0][0];
        #pragma unroll
        for (int nt = 0; nt < 8; nt++) {
            v16bf b = load_fragB(wbuf + (nt * 16 + m) * 32 + kg * 8);
            acc1[nt] = __builtin_amdgcn_wmma_f32_16x16x32_bf16(
                false, a, false, b, (short)0, acc1[nt], false, false);
        }
        if (idx < 15) { wait_async0(); __syncthreads(); }
    }
    #pragma unroll
    for (int nt = 0; nt < 8; nt++) {
        int col = nt * 16 + m;
        float bb = b1[col];
        #pragma unroll
        for (int r = 0; r < 8; r++) {
            float v = acc1[nt][r] + bb;
            relay1[wv][kg * 8 + r][col] = (__bf16)(v > 0.0f ? v : 0.0f);
        }
    }
    __syncthreads();

    // ---- layer 2: [16,128] x [128,64], B from LDS-resident W2t ----
    v8f acc2[4];
    #pragma unroll
    for (int i = 0; i < 4; i++) acc2[i] = (v8f){};
    #pragma unroll
    for (int k0 = 0; k0 < 128; k0 += 32) {
        v16bf a = load_fragA(&relay1[wv][m][k0 + kg * 8]);
        #pragma unroll
        for (int nt = 0; nt < 4; nt++) {
            v16bf b = load_fragB(&w2s[nt * 16 + m][k0 + kg * 8]);
            acc2[nt] = __builtin_amdgcn_wmma_f32_16x16x32_bf16(
                false, a, false, b, (short)0, acc2[nt], false, false);
        }
    }
    #pragma unroll
    for (int nt = 0; nt < 4; nt++) {
        int col = nt * 16 + m;
        float bb = b2[col];
        #pragma unroll
        for (int r = 0; r < 8; r++) {
            float v = acc2[nt][r] + bb;
            relay2[wv][kg * 8 + r][col] = (__bf16)(v > 0.0f ? v : 0.0f);
        }
    }
    __syncthreads();

    // ---- layer 3: 64 -> 1, sigmoid*4+1 ----
    if (lane < 16 && (ebase + lane) < E) {
        float sacc = b3[0];
        #pragma unroll
        for (int j = 0; j < 64; j++) sacc += (float)relay2[wv][lane][j] * W3[j];
        out[ebase + lane] = 4.0f / (1.0f + __expf(-sacc)) + 1.0f;
    }
}

// ---------------------------------------------------------------------------
extern "C" void kernel_launch(void* const* d_in, const int* in_sizes, int n_in,
                              void* d_out, int out_size, void* d_ws, size_t ws_size,
                              hipStream_t stream) {
    const long long* ei = (const long long*)d_in[0];   // edge_index int64 [2,E]
    // d_in[1] = edge_time (unused by the reference forward)
    const float* x   = (const float*)d_in[2];
    const float* mem = (const float*)d_in[3];
    const float* Wq = (const float*)d_in[4];  const float* bq = (const float*)d_in[5];
    const float* Wk = (const float*)d_in[6];  const float* bk = (const float*)d_in[7];
    const float* Wv = (const float*)d_in[8];  const float* bv = (const float*)d_in[9];
    const float* Ws = (const float*)d_in[10]; const float* bs = (const float*)d_in[11];
    const float* W1 = (const float*)d_in[12]; const float* b1 = (const float*)d_in[13];
    const float* W2 = (const float*)d_in[14]; const float* b2 = (const float*)d_in[15];
    const float* W3 = (const float*)d_in[16]; const float* b3 = (const float*)d_in[17];
    (void)n_in; (void)out_size; (void)ws_size;

    const int N = in_sizes[2] / HID;   // 100000
    const int E = in_sizes[0] / 2;     // 400000

    // ---- workspace bump allocator (256B aligned) ----
    char* p = (char*)d_ws;
    auto alloc = [&](size_t bytes) -> void* {
        void* r = (void*)p;
        p += (bytes + 255) & ~(size_t)255;
        return r;
    };
    __bf16* hb     = (__bf16*)alloc((size_t)N * HID * 2);       // h bf16
    __bf16* qb     = (__bf16*)alloc((size_t)N * HD * 2);
    __bf16* kb     = (__bf16*)alloc((size_t)N * HD * 2);
    __bf16* vb     = (__bf16*)alloc((size_t)N * HD * 2);
    float*  accb   = (float*) alloc((size_t)N * HD * 4);        // skip + agg (fp32)
    __bf16* outb   = (__bf16*)alloc((size_t)N * HD * 2);
    float*  scores = (float*) alloc((size_t)E * HEADS * 4);
    float*  smax   = (float*) alloc((size_t)N * HEADS * 4);
    float*  denom  = (float*) alloc((size_t)N * HEADS * 4);
    __bf16* Wt     = (__bf16*)alloc((size_t)4 * HD * HID * 2);  // q/k/v/skip transposed
    __bf16* W1t    = (__bf16*)alloc((size_t)128 * 512 * 2);
    __bf16* W2t    = (__bf16*)alloc((size_t)64 * 128 * 2);

    const int T = 256;

    // weight transpose/convert (tiny, L2 resident afterwards)
    k_transpose_cvt<<<(HID * HD + T - 1) / T, T, 0, stream>>>(Wq, Wt + 0 * HD * HID, HID, HD);
    k_transpose_cvt<<<(HID * HD + T - 1) / T, T, 0, stream>>>(Wk, Wt + 1 * HD * HID, HID, HD);
    k_transpose_cvt<<<(HID * HD + T - 1) / T, T, 0, stream>>>(Wv, Wt + 2 * HD * HID, HID, HD);
    k_transpose_cvt<<<(HID * HD + T - 1) / T, T, 0, stream>>>(Ws, Wt + 3 * HD * HID, HID, HD);
    k_transpose_cvt<<<(512 * 128 + T - 1) / T, T, 0, stream>>>(W1, W1t, 512, 128);
    k_transpose_cvt<<<(128 * 64 + T - 1) / T, T, 0, stream>>>(W2, W2t, 128, 64);

    k_init<<<(N * HEADS + T - 1) / T, T, 0, stream>>>(smax, denom, N * HEADS);

    // h = memory + x  (bf16)
    {
        int n = N * HID;
        k_addcvt<<<(n + T - 1) / T, T, 0, stream>>>(mem, x, hb, n);
    }

    // q/k/v/skip projections via WMMA (A fragments register-resident, 16 col tiles)
    {
        dim3 grid((N + 15) / 16, 4);
        k_node_gemm<<<grid, 32, 0, stream>>>(hb, Wt, bq, bk, bv, bs, qb, kb, vb, accb, N);
    }

    // attention softmax over in-edges of dst
    k_scores<<<(E * HEADS + T - 1) / T, T, 0, stream>>>(ei, qb, kb, scores, smax, E);
    k_expnorm<<<(E * HEADS + T - 1) / T, T, 0, stream>>>(ei, scores, smax, denom, E);
    k_scatter_v<<<((size_t)E * (HD / 4) + T - 1) / T, T, 0, stream>>>(ei, vb, scores, denom, accb, E);

    // out = agg + skip -> bf16 for MLP A-operands
    k_cvt_out<<<((size_t)N * HD + T - 1) / T, T, 0, stream>>>(accb, outb, N * HD);

    // fused rating MLP: 4 waves/block, async-staged weights, 144 WMMA per wave
    k_edge_mlp<<<(E + 63) / 64, 128, 0, stream>>>(ei, outb, W1t, b1, W2t, b2, W3, b3,
                                                  (float*)d_out, E);
}
